// DomainMLP_3332894622574
// MI455X (gfx1250) — compile-verified, matrix operand order checked by reading, not run on hardware
//
#include <hip/hip_runtime.h>

#define BB   65536
#define INN  784
#define INP  800          // 784 padded up to a multiple of 32
#define HN   768
#define NE   192          // D*E = 3*64
#define KPAD 40           // 80B LDS row stride: 16B aligned, conflict-free across 16 lanes

typedef unsigned short u16;
typedef __attribute__((ext_vector_type(16))) __bf16 v16bf;
typedef __attribute__((ext_vector_type(8)))  float  v8f;

#if defined(__HIP_DEVICE_COMPILE__) && __has_builtin(__builtin_amdgcn_global_load_async_to_lds_b128)
#define USE_ASYNC 1
#else
#define USE_ASYNC 0
#endif

__device__ inline u16 f2bf(float f) {
  unsigned int u = __float_as_uint(f);
  u += 0x7FFFu + ((u >> 16) & 1u);           // RNE
  return (u16)(u >> 16);
}
__device__ inline float bf2f(u16 u) { return __uint_as_float(((unsigned int)u) << 16); }

__device__ inline v8f wmma_bf16(v16bf a, v16bf b, v8f c) {
  return __builtin_amdgcn_wmma_f32_16x16x32_bf16(false, a, false, b, (short)0, c, false, false);
}

// Per-lane fragment load from an LDS slab staged [row][KPAD] (K contiguous).
// A frag row = m; B frag row = n (B is stored K-major per output column).
// Element i -> k = (i>>3)*16 + lanehalf*8 + (i&7)   (ISA 7.12.2, 16-bit 16x32)
__device__ inline v16bf load_frag(const u16* __restrict__ S, int row, int lh) {
  const u16* p = S + row * KPAD + lh * 8;      // byte off = row*80 + lh*16 : 16B aligned
  union { uint4 u[2]; v16bf v; } cvt;
  cvt.u[0] = *(const uint4*)(p);
  cvt.u[1] = *(const uint4*)(p + 16);
  return cvt.v;
}

#if USE_ASYNC
typedef int v4i __attribute__((vector_size(16)));
typedef __attribute__((address_space(1))) v4i as1_v4i;   // global
typedef __attribute__((address_space(3))) v4i as3_v4i;   // LDS
#endif

__device__ inline void async_cp16(const u16* gp, u16* sp) {
#if USE_ASYNC
  __builtin_amdgcn_global_load_async_to_lds_b128((as1_v4i*)gp, (as3_v4i*)sp, 0, 0);
#else
  *(uint4*)sp = *(const uint4*)gp;
#endif
}

__device__ inline void wait_async_all() {
#if USE_ASYNC
#if __has_builtin(__builtin_amdgcn_s_wait_asynccnt)
  __builtin_amdgcn_s_wait_asynccnt(0);
#else
  asm volatile("s_wait_asynccnt 0x0" ::: "memory");
#endif
#endif
}

// Copy rows [r0, r0+ROWS) x cols [k0, k0+32) of a K-contiguous matrix (ld) into s[ROWS][KPAD]
template<int ROWS>
__device__ inline void stage_rows(const u16* __restrict__ g, u16* __restrict__ s,
                                  size_t r0, int k0, int ld, int tid) {
#pragma unroll
  for (int c = tid; c < ROWS * 4; c += 256) {
    int r = c >> 2, kg = c & 3;
    const u16* gp = g + (r0 + r) * (size_t)ld + k0 + kg * 8;
    u16* sp = s + r * KPAD + kg * 8;
    async_cp16(gp, sp);
  }
}

// Double-buffered WMMA GEMM core. A: [.. x lda] rows at row0; Bt: K-major, rows = out cols at col0.
template<int BM, int BN, int WM, int WN, int KT, int MF, int NF>
__device__ inline void gemm_core(const u16* __restrict__ A, const u16* __restrict__ Bt,
                                 size_t row0, size_t col0, int lda, int ldb,
                                 u16* __restrict__ As, u16* __restrict__ Bs,
                                 v8f (&acc)[MF][NF]) {
  const int tid = threadIdx.x, lane = tid & 31, wave = tid >> 5;
  const int wm = wave / WN, wn = wave % WN;
  const int lh = lane >> 4, ml = lane & 15;
#pragma unroll
  for (int i = 0; i < MF; ++i)
#pragma unroll
    for (int j = 0; j < NF; ++j)
#pragma unroll
      for (int r = 0; r < 8; ++r) acc[i][j][r] = 0.f;

  stage_rows<BM>(A,  As, row0, 0, lda, tid);
  stage_rows<BN>(Bt, Bs, col0, 0, ldb, tid);
  wait_async_all();
  __syncthreads();

  int buf = 0;
  for (int k0 = 0; k0 < KT; k0 += 32, buf ^= 1) {
    if (k0 + 32 < KT) {                       // prefetch next K-slab into other buffer
      stage_rows<BM>(A,  As + (buf ^ 1) * BM * KPAD, row0, k0 + 32, lda, tid);
      stage_rows<BN>(Bt, Bs + (buf ^ 1) * BN * KPAD, col0, k0 + 32, ldb, tid);
    }
    const u16* Ab = As + buf * BM * KPAD;
    const u16* Bb = Bs + buf * BN * KPAD;
    v16bf af[MF];
#pragma unroll
    for (int mf = 0; mf < MF; ++mf)
      af[mf] = load_frag(Ab, wm * (BM / WM) + mf * 16 + ml, lh);
#pragma unroll
    for (int nf = 0; nf < NF; ++nf) {
      v16bf bf_ = load_frag(Bb, wn * (BN / WN) + nf * 16 + ml, lh);
#pragma unroll
      for (int mf = 0; mf < MF; ++mf)
        acc[mf][nf] = wmma_bf16(af[mf], bf_, acc[mf][nf]);
    }
    wait_async_all();
    __syncthreads();
  }
}

// ---------------- conversion / init kernels ----------------
__global__ void k_cvt_padcols(const float* __restrict__ src, u16* __restrict__ dst,
                              int rows, int cols, int dcols) {
  size_t i = (size_t)blockIdx.x * blockDim.x + threadIdx.x;
  if (i >= (size_t)rows * dcols) return;
  int c = (int)(i % dcols); size_t r = i / dcols;
  dst[i] = f2bf(c < cols ? src[r * cols + c] : 0.f);
}
// dst[n][k] = src[k][n] (bf16), k padded [0,krows)->kpad with zeros
__global__ void k_cvt_T(const float* __restrict__ src, u16* __restrict__ dst,
                        int krows, int ncols, int kpad) {
  size_t i = (size_t)blockIdx.x * blockDim.x + threadIdx.x;
  if (i >= (size_t)ncols * kpad) return;
  int k = (int)(i % kpad); size_t n = i / kpad;
  dst[i] = f2bf(k < krows ? src[(size_t)k * ncols + n] : 0.f);
}
// dst[d*64+e][h] = Wa[d][h][e]   (K-major expert weights, all 3 experts)
__global__ void k_cvt_waT(const float* __restrict__ src, u16* __restrict__ dst) {
  int i = blockIdx.x * blockDim.x + threadIdx.x;
  if (i >= NE * HN) return;
  int h = i % HN, de = i / HN, d = de >> 6, e = de & 63;
  dst[i] = f2bf(src[((size_t)d * HN + h) * 64 + e]);
}
__global__ void k_zero(float* p, int n) {
  int i = blockIdx.x * blockDim.x + threadIdx.x;
  if (i < n) p[i] = 0.f;
}
__global__ void k_dom_init(float* __restrict__ domws, const float* __restrict__ bd2) {
  int i = blockIdx.x * blockDim.x + threadIdx.x;
  if (i < BB * 3) domws[i] = bd2[i % 3];
}

// ---------------- GEMM1: h = x@W1 + b1, + BN column stats ----------------
__global__ __launch_bounds__(256) void k_gemm1(const u16* __restrict__ Ab,
    const u16* __restrict__ Bt, const float* __restrict__ bias,
    u16* __restrict__ Hout, float* __restrict__ stats) {
  __shared__ __align__(16) u16 As[2 * 128 * KPAD];
  __shared__ __align__(16) u16 Bs[2 * 128 * KPAD];
  __shared__ float csum[128], csq[128];
  int tid = threadIdx.x, lane = tid & 31, wave = tid >> 5;
  int wm = wave >> 1, wn = wave & 1;
  size_t row0 = (size_t)blockIdx.y * 128; size_t col0 = (size_t)blockIdx.x * 128;
  int lh = lane >> 4, ml = lane & 15;
  v8f acc[2][4];
  gemm_core<128, 128, 4, 2, INP, 2, 4>(Ab, Bt, row0, col0, INP, INP, As, Bs, acc);

  if (tid < 128) { csum[tid] = 0.f; csq[tid] = 0.f; }
  __syncthreads();
#pragma unroll
  for (int mf = 0; mf < 2; ++mf)
#pragma unroll
    for (int nf = 0; nf < 4; ++nf) {
      int n = wn * 64 + nf * 16 + ml;
      float bv = bias[col0 + n];
      float s = 0.f, q = 0.f;
#pragma unroll
      for (int r = 0; r < 8; ++r) {
        int m = wm * 32 + mf * 16 + lh * 8 + r;
        float v = acc[mf][nf][r] + bv;
        Hout[(row0 + m) * HN + col0 + n] = f2bf(v);
        s += v; q += v * v;
      }
      atomicAdd(&csum[n], s); atomicAdd(&csq[n], q);
    }
  __syncthreads();
  if (tid < 128) {
    atomicAdd(&stats[col0 + tid], csum[tid]);
    atomicAdd(&stats[HN + col0 + tid], csq[tid]);
  }
}

// ---------------- BN finalize + in-place BN/ReLU ----------------
__global__ void k_bn_fin(const float* __restrict__ stats, const float* __restrict__ gamma,
                         const float* __restrict__ beta, float* __restrict__ ss) {
  int h = blockIdx.x * blockDim.x + threadIdx.x;
  if (h >= HN) return;
  float mu = stats[h] * (1.0f / BB);
  float var = stats[HN + h] * (1.0f / BB) - mu * mu;
  float sc = gamma[h] * rsqrtf(fmaxf(var, 0.f) + 1e-5f);
  ss[h] = sc; ss[HN + h] = beta[h] - mu * sc;
}
__global__ void k_bn_relu(u16* __restrict__ hb, const float* __restrict__ ss) {
  size_t i = (size_t)blockIdx.x * blockDim.x + threadIdx.x;   // over BB*HN/8
  if (i >= (size_t)BB * HN / 8) return;
  int h0 = (int)((i * 8) % HN);
  uint4 v = *(uint4*)(hb + i * 8);
  u16* pe = (u16*)&v;
#pragma unroll
  for (int j = 0; j < 8; ++j) {
    float f = bf2f(pe[j]) * ss[h0 + j] + ss[HN + h0 + j];
    pe[j] = f2bf(fmaxf(f, 0.f));
  }
  *(uint4*)(hb + i * 8) = v;
}

// ---- GEMM2: t=relu(feat@Wd1+bd1) kept in LDS; fused t@Wd2 -> dom_out atomics ----
__global__ __launch_bounds__(256) void k_gemm2(const u16* __restrict__ F,
    const u16* __restrict__ Bt, const float* __restrict__ bias,
    const float* __restrict__ Wd2, float* __restrict__ domws) {
  __shared__ __align__(16) u16 As[2 * 128 * KPAD];
  __shared__ __align__(16) u16 Bs[2 * 128 * KPAD];
  __shared__ __align__(16) u16 Ts[128 * 130];
  int tid = threadIdx.x, lane = tid & 31, wave = tid >> 5;
  int wm = wave >> 1, wn = wave & 1;
  size_t row0 = (size_t)blockIdx.y * 128; size_t col0 = (size_t)blockIdx.x * 128;
  int lh = lane >> 4, ml = lane & 15;
  v8f acc[2][4];
  gemm_core<128, 128, 4, 2, HN, 2, 4>(F, Bt, row0, col0, HN, HN, As, Bs, acc);

#pragma unroll
  for (int mf = 0; mf < 2; ++mf)
#pragma unroll
    for (int nf = 0; nf < 4; ++nf) {
      int n = wn * 64 + nf * 16 + ml;
      float bv = bias[col0 + n];
#pragma unroll
      for (int r = 0; r < 8; ++r) {
        int m = wm * 32 + mf * 16 + lh * 8 + r;
        Ts[m * 130 + n] = f2bf(fmaxf(acc[mf][nf][r] + bv, 0.f));
      }
    }
  __syncthreads();
  if (tid < 128) {                      // column-reduce tile against Wd2 slice
    float s0 = 0.f, s1 = 0.f, s2 = 0.f;
    for (int n = 0; n < 128; ++n) {
      float tv = bf2f(Ts[tid * 130 + n]);
      const float* w = Wd2 + (col0 + n) * 3;
      s0 += tv * w[0]; s1 += tv * w[1]; s2 += tv * w[2];
    }
    float* dp = domws + (row0 + tid) * 3;
    atomicAdd(dp + 0, s0); atomicAdd(dp + 1, s1); atomicAdd(dp + 2, s2);
  }
}

// ---- GEMM3: hid=relu(feat@Wa_cat+ba) in LDS; fused argmax-route + expert head ----
__global__ __launch_bounds__(256) void k_gemm3(const u16* __restrict__ F,
    const u16* __restrict__ Bt, const float* __restrict__ ba,
    const float* __restrict__ domws, const float* __restrict__ Wb,
    const float* __restrict__ bb, float* __restrict__ dout) {
  __shared__ __align__(16) u16 As[2 * 64 * KPAD];
  __shared__ __align__(16) u16 Bs[2 * NE * KPAD];
  __shared__ __align__(16) u16 Hs[64 * 194];
  int tid = threadIdx.x, lane = tid & 31, wave = tid >> 5;
  int wm = wave >> 2, wn = wave & 3;               // 2(M) x 4(N), wave tile 32x48
  size_t row0 = (size_t)blockIdx.x * 64;
  int lh = lane >> 4, ml = lane & 15;
  v8f acc[2][3];
  gemm_core<64, NE, 2, 4, HN, 2, 3>(F, Bt, row0, 0, HN, HN, As, Bs, acc);

#pragma unroll
  for (int mf = 0; mf < 2; ++mf)
#pragma unroll
    for (int nf = 0; nf < 3; ++nf) {
      int n = wn * 48 + nf * 16 + ml;
      float bv = ba[n];
#pragma unroll
      for (int r = 0; r < 8; ++r) {
        int m = wm * 32 + mf * 16 + lh * 8 + r;
        Hs[m * 194 + n] = f2bf(fmaxf(acc[mf][nf][r] + bv, 0.f));
      }
    }
  __syncthreads();
  if (tid < 64) {
    size_t gb = row0 + tid;
    const float* dv = domws + gb * 3;
    float d0 = dv[0], d1 = dv[1], d2 = dv[2];
    int pred = 0; float best = d0;
    if (d1 > best) { pred = 1; best = d1; }
    if (d2 > best) { pred = 2; }
    float o[10];
#pragma unroll
    for (int c = 0; c < 10; ++c) o[c] = bb[pred * 10 + c];
    for (int e = 0; e < 64; ++e) {
      float hv = bf2f(Hs[tid * 194 + pred * 64 + e]);
      const float* w = Wb + (size_t)(pred * 64 + e) * 10;
#pragma unroll
      for (int c = 0; c < 10; ++c) o[c] += hv * w[c];
    }
    float* op = dout + gb * 10;
#pragma unroll
    for (int c = 0; c < 10; ++c) op[c] = o[c];
    float* dp = dout + (size_t)BB * 10 + gb * 3;
    dp[0] = d0; dp[1] = d1; dp[2] = d2;
  }
}

extern "C" void kernel_launch(void* const* d_in, const int* in_sizes, int n_in,
                              void* d_out, int out_size, void* d_ws, size_t ws_size,
                              hipStream_t stream) {
  const float* x     = (const float*)d_in[0];
  const float* W1    = (const float*)d_in[1];
  const float* b1    = (const float*)d_in[2];
  const float* gamma = (const float*)d_in[3];
  const float* beta  = (const float*)d_in[4];
  const float* Wd1   = (const float*)d_in[5];
  const float* bd1   = (const float*)d_in[6];
  const float* Wd2   = (const float*)d_in[7];
  const float* bd2   = (const float*)d_in[8];
  const float* Wa    = (const float*)d_in[9];
  const float* ba    = (const float*)d_in[10];
  const float* Wb    = (const float*)d_in[11];
  const float* bb    = (const float*)d_in[12];
  float* dout = (float*)d_out;

  char* ws = (char*)d_ws;
  size_t off = 0;
  auto alloc = [&](size_t bytes) -> void* {
    off = (off + 255) & ~(size_t)255;
    void* p = ws + off; off += bytes; return p;
  };
  u16*   xb    = (u16*)  alloc((size_t)BB * INP * 2);    // x bf16, K padded to 800
  u16*   hb    = (u16*)  alloc((size_t)BB * HN * 2);     // h -> feat (in place)
  u16*   W1T   = (u16*)  alloc((size_t)HN * INP * 2);    // W1^T  [768][800]
  u16*   Wd1T  = (u16*)  alloc((size_t)HN * HN * 2);     // Wd1^T [768][768]
  u16*   WaT   = (u16*)  alloc((size_t)NE * HN * 2);     // Wa_cat^T [192][768]
  float* stats = (float*)alloc(2 * HN * 4);
  float* ss    = (float*)alloc(2 * HN * 4);
  float* domws = (float*)alloc((size_t)BB * 3 * 4);
  if (off > ws_size) return;   // workspace too small: bail rather than corrupt

  {
    size_t n = (size_t)BB * INP;
    k_cvt_padcols<<<dim3((unsigned)((n + 255) / 256)), 256, 0, stream>>>(x, xb, BB, INN, INP);
  }
  k_cvt_T<<<(HN * INP + 255) / 256, 256, 0, stream>>>(W1, W1T, INN, HN, INP);
  k_cvt_T<<<(HN * HN + 255) / 256, 256, 0, stream>>>(Wd1, Wd1T, HN, HN, HN);
  k_cvt_waT<<<(NE * HN + 255) / 256, 256, 0, stream>>>(Wa, WaT);
  k_zero<<<(2 * HN + 255) / 256, 256, 0, stream>>>(stats, 2 * HN);
  k_dom_init<<<(BB * 3 + 255) / 256, 256, 0, stream>>>(domws, bd2);

  k_gemm1<<<dim3(HN / 128, BB / 128), 256, 0, stream>>>(xb, W1T, b1, hb, stats);
  k_bn_fin<<<(HN + 255) / 256, 256, 0, stream>>>(stats, gamma, beta, ss);
  {
    size_t n = (size_t)BB * HN / 8;
    k_bn_relu<<<dim3((unsigned)((n + 255) / 256)), 256, 0, stream>>>(hb, ss);
  }
  k_gemm2<<<dim3(HN / 128, BB / 128), 256, 0, stream>>>(hb, Wd1T, bd1, Wd2, domws);
  k_gemm3<<<dim3(BB / 64), 256, 0, stream>>>(hb, WaT, ba, domws, Wb, bb, dout);
}
